// PConvLinear_4097398800823
// MI455X (gfx1250) — matrix operand order repaired and checked by reading, not compiled, not run on hardware
//
#include <hip/hip_runtime.h>
#include <hip/hip_bf16.h>

typedef __attribute__((ext_vector_type(2)))  float        v2f;
typedef __attribute__((ext_vector_type(8)))  float        v8f;
typedef __attribute__((ext_vector_type(16))) __bf16       v16bf;
typedef __attribute__((ext_vector_type(2)))  int          i32x2;
typedef __attribute__((ext_vector_type(4)))  unsigned int u32x4;
typedef __attribute__((ext_vector_type(8)))  unsigned int u32x8;

#define NPTS   60000
#define NBRK   16
#define CIN    64
#define CADD   3
#define CMID   16
#define FIN    1072   // (CIN+CADD)*CMID
#define FPAD   1088   // FIN padded up to a multiple of 32 (bf16 WMMA K-step)
#define OUTF   128
#define MT     32     // points per workgroup
#define TOTPTS (2 * NPTS)

__device__ __forceinline__ unsigned bf16_rne(float x) {
    unsigned u = __float_as_uint(x);
    return (u + 0x7fffu + ((u >> 16) & 1u)) >> 16;   // round-to-nearest-even
}

__device__ __forceinline__ v16bf mk_v16bf(u32x4 lo, u32x4 hi) {
    u32x8 u;
    u[0] = lo[0]; u[1] = lo[1]; u[2] = lo[2]; u[3] = lo[3];
    u[4] = hi[0]; u[5] = hi[1]; u[6] = hi[2]; u[7] = hi[3];
    return __builtin_bit_cast(v16bf, u);
}

// ---- Prepass: linear_weight f32 [128][1072] -> bf16 [128][1088] (zero-padded K) ----
__global__ void cvt_weight_kernel(const float* __restrict__ w, unsigned int* __restrict__ wbf) {
    int id = blockIdx.x * 256 + threadIdx.x;           // one dword = 2 bf16
    if (id >= OUTF * (FPAD / 2)) return;
    int o = id / (FPAD / 2);
    int f = (id % (FPAD / 2)) * 2;
    unsigned lo = (f     < FIN) ? bf16_rne(w[o * FIN + f])     : 0u;
    unsigned hi = (f + 1 < FIN) ? bf16_rne(w[o * FIN + f + 1]) : 0u;
    wbf[id] = lo | (hi << 16);
}

// ---- Main fused kernel: gather + PConv (f32 WMMA) + linear (bf16 WMMA) ----
__global__ __launch_bounds__(256, 1) void pconv_linear_kernel(
    const float* __restrict__ xf,  const int* __restrict__ nbr,
    const float* __restrict__ wn,  const float* __restrict__ af,
    const unsigned short* __restrict__ wbf, const float* __restrict__ bias,
    float* __restrict__ out)
{
    __shared__ __align__(16) unsigned short s_pconv[MT * FPAD];  // 69,632 B bf16

    const int tid  = threadIdx.x;
    const int lane = tid & 31;
    const int wv   = tid >> 5;     // wave id 0..7
    const int cl   = lane & 15;    // column within half-wave
    const int half = lane >> 4;    // 0 / 1
    const int pt0  = blockIdx.x * MT;

    // zero the K-padding region [FIN, FPAD) for all MT points (256 dwords, one per thread)
    {
        int p = tid >> 3, j = tid & 7;
        *(unsigned int*)&s_pconv[p * FPAD + FIN + j * 2] = 0u;
    }

    // ========= Stage A: pconv[p][c][m] via V_WMMA_F32_16X16X4_F32, 4 points/wave =========
    const unsigned clc = (unsigned)((cl < CADD) ? cl : (CADD - 1));  // clamped column for `af`
    for (int i = 0; i < 4; ++i) {
        const int p = wv * 4 + i;
        const int n = pt0 + p;                              // global flattened point
        const unsigned fbase = (n >= NPTS) ? (unsigned)NPTS * CIN : 0u;  // batch base (uniform)

        v8f acc[5];
        #pragma unroll
        for (int t = 0; t < 5; ++t) acc[t] = (v8f)0.0f;

        #pragma unroll
        for (int ks = 0; ks < 4; ++ks) {
            const int k0 = ks * 4 + half * 2;               // this lane's two K values: k0, k0+1
            const i32x2 ii = *(const i32x2*)&nbr[n * NBRK + k0];   // b64: idx pair
            v2f a;                                          // A = w^T : A[m][k] = weightnet[n][k][m]
            {
                const unsigned wo = (unsigned)n * (NBRK * CMID) + (unsigned)k0 * CMID + (unsigned)cl;
                a.x = wn[wo];
                a.y = wn[wo + CMID];
            }
            const unsigned off0 = fbase + ((unsigned)ii[0] << 6) + (unsigned)cl;  // row of neighbor k0
            const unsigned off1 = fbase + ((unsigned)ii[1] << 6) + (unsigned)cl;  // row of neighbor k0+1
            #pragma unroll
            for (int nt = 0; nt < 4; ++nt) {                // gathered-feature tiles (c < 64)
                v2f bb;
                bb.x = xf[off0 + nt * 16];
                bb.y = xf[off1 + nt * 16];
                acc[nt] = __builtin_amdgcn_wmma_f32_16x16x4_f32(
                    false, a, false, bb, (short)0, acc[nt], false, false);
            }
            // additional-feature tile (c = 64..66): branchless clamped load + select
            {
                const unsigned ao = (unsigned)n * (NBRK * CADD) + (unsigned)k0 * CADD + clc;
                const float a0 = af[ao];
                const float a1 = af[ao + CADD];
                v2f bb;
                bb.x = (cl < CADD) ? a0 : 0.0f;
                bb.y = (cl < CADD) ? a1 : 0.0f;
                acc[4] = __builtin_amdgcn_wmma_f32_16x16x4_f32(
                    false, a, false, bb, (short)0, acc[4], false, false);
            }
        }

        // D layout: lane = c (N), element r = m = half*8 + r; f = c*16 + m is contiguous
        #pragma unroll
        for (int nt = 0; nt < 5; ++nt) {
            const int c = nt * 16 + cl;
            if (c < CIN + CADD) {
                u32x4 pk;
                #pragma unroll
                for (int r = 0; r < 4; ++r)
                    pk[r] = bf16_rne(acc[nt][2 * r]) | (bf16_rne(acc[nt][2 * r + 1]) << 16);
                *(u32x4*)&s_pconv[p * FPAD + c * CMID + half * 8] = pk;
            }
        }
    }

    __syncthreads();

    // ========= Stage B: [32 pts x 1088] @ [1088 x 128] via V_WMMA_F32_16X16X32_BF16 =====
    const int o = wv * 16 + cl;                             // this lane's output channel
    const float bv = bias[o];
    const unsigned wrow = (unsigned)o * FPAD + (unsigned)(half * 8);  // 32-bit bf16-offset

    v8f acc0 = (v8f)0.0f, acc1 = (v8f)0.0f;
    #pragma unroll 2
    for (int ks = 0; ks < FPAD / 32; ++ks) {                // 34 K-steps
        const int kb = ks * 32;
        if (ks + 8 < FPAD / 32)                             // uniform-guarded weight prefetch
            __builtin_prefetch(wbf + wrow + kb + 8 * 32, 0, 0);
        const int kA = kb + half * 8;                       // A frag: K kA..kA+7, kA+16..kA+23
        u32x4 a0lo = *(const u32x4*)&s_pconv[cl * FPAD + kA];
        u32x4 a0hi = *(const u32x4*)&s_pconv[cl * FPAD + kA + 16];
        u32x4 a1lo = *(const u32x4*)&s_pconv[(cl + 16) * FPAD + kA];
        u32x4 a1hi = *(const u32x4*)&s_pconv[(cl + 16) * FPAD + kA + 16];
        u32x4 blo  = *(const u32x4*)(wbf + wrow + kb);      // B frag: K kb+8h..+7, kb+16+8h..+7
        u32x4 bhi  = *(const u32x4*)(wbf + wrow + kb + 16);
        v16bf A0 = mk_v16bf(a0lo, a0hi);
        v16bf A1 = mk_v16bf(a1lo, a1hi);
        v16bf Bf = mk_v16bf(blo, bhi);
        acc0 = __builtin_amdgcn_wmma_f32_16x16x32_bf16(false, A0, false, Bf, (short)0, acc0, false, false);
        acc1 = __builtin_amdgcn_wmma_f32_16x16x32_bf16(false, A1, false, Bf, (short)0, acc1, false, false);
    }

    #pragma unroll
    for (int r = 0; r < 8; ++r) {
        const int m = half * 8 + r;
        out[(unsigned)(pt0 + m) * OUTF + (unsigned)o]      = acc0[r] + bv;
        out[(unsigned)(pt0 + 16 + m) * OUTF + (unsigned)o] = acc1[r] + bv;
    }
}

extern "C" void kernel_launch(void* const* d_in, const int* in_sizes, int n_in,
                              void* d_out, int out_size, void* d_ws, size_t ws_size,
                              hipStream_t stream) {
    const float* xf  = (const float*)d_in[0];   // input_features  [2,60000,64]
    const int*   nbr = (const int*)d_in[1];     // neighbor_inds   [2,60000,16]
    const float* wn  = (const float*)d_in[2];   // weightnet       [2,60000,16,16]
    const float* af  = (const float*)d_in[3];   // additional      [2,60000,16,3]
    const float* lw  = (const float*)d_in[4];   // linear_weight   [128,1072]
    const float* lb  = (const float*)d_in[5];   // linear_bias     [128]

    // d_ws: bf16 weight [128][1088] = 278,528 bytes
    unsigned int* wbf = (unsigned int*)d_ws;
    const int cvt_dwords = OUTF * (FPAD / 2);
    cvt_weight_kernel<<<(cvt_dwords + 255) / 256, 256, 0, stream>>>(lw, wbf);

    pconv_linear_kernel<<<TOTPTS / MT, 256, 0, stream>>>(
        xf, nbr, wn, af, (const unsigned short*)d_ws, lb, (float*)d_out);
}
